// MultiHeadAttention_6339371729143
// MI455X (gfx1250) — matrix operand order repaired and checked by reading, not compiled, not run on hardware
//
#include <hip/hip_runtime.h>

// ---------------------------------------------------------------------------
// MHA forward for MI455X (gfx1250, wave32, WMMA).
//   B=2, S=2048, H=1024, heads=16, d_k=64.  ~60 GFLOP / ~66MB working set.
// All six GEMMs run on v_wmma_f32_16x16x32_bf16. One upfront elementwise pass
// converts activations+weights to bf16 so every GEMM inner loop is pure
// global_load_b128 + v_wmma (no per-tile f32->bf16 VALU work). The whole
// working set is L2-resident (192MB), so register-blocked GEMMs with direct
// global loads are sufficient. Softmax row-sums are computed by an extra
// WMMA against an all-ones B fragment (no cross-lane reductions for l);
// only the row-max needs shuffles, amortized over 64-key steps.
// ---------------------------------------------------------------------------

#define B_    2
#define S_    2048
#define HDIM  1024
#define NH    16
#define DK    64
#define PSTR  72   // LDS P-tile row stride (bf16 elems): 144B, 16B-aligned

typedef __attribute__((ext_vector_type(16))) __bf16 v16bf;
typedef __attribute__((ext_vector_type(8)))  float  v8f;

static __device__ __forceinline__ v8f wmma_bf16(v16bf a, v16bf b, v8f c) {
  return __builtin_amdgcn_wmma_f32_16x16x32_bf16(
      false, a, false, b, (short)0, c, false, false);
}

// A-fragment, 16-bit 16x32 (ISA 7.12.2): lane m = lane&15.
// lanes 0-15 hold K {0..7, 16..23}; lanes 16-31 hold K {8..15, 24..31}.
static __device__ __forceinline__ v16bf afrag_bf16(const __bf16* q) {
  const int koff = (threadIdx.x & 16) ? 8 : 0;
  union { uint4 u[2]; v16bf v; } x;
  x.u[0] = *(const uint4*)(q + koff);
  x.u[1] = *(const uint4*)(q + koff + 16);
  return x.v;
}

// B-fragment, 16-bit 32x16: lane n = lane&15.
// lanes 0-15 hold K {0..7, 16..23}; lanes 16-31 hold K {8..15, 24..31}.
static __device__ __forceinline__ v16bf bfrag_bf16(const __bf16* q) {
  const int koff = (threadIdx.x & 16) ? 8 : 0;
  union { uint4 u[2]; v16bf v; } x;
  x.u[0] = *(const uint4*)(q + koff);
  x.u[1] = *(const uint4*)(q + koff + 16);
  return x.v;
}

// ---------------------------------------------------------------------------
// One-shot f32 -> bf16 conversion (memory-bound; runs once per tensor).
// Each thread converts 8 contiguous elements: 2x b128 in, 1x b128 out.
// ---------------------------------------------------------------------------
__global__ __launch_bounds__(256)
void cvt_bf16_kernel(const float* __restrict__ in, __bf16* __restrict__ out, int n) {
  const int i = (blockIdx.x * 256 + threadIdx.x) * 8;
  if (i >= n) return;
  const float4 a = *(const float4*)(in + i);
  const float4 b = *(const float4*)(in + i + 4);
  union { uint4 u; __bf16 h[8]; } x;
  x.h[0] = (__bf16)a.x; x.h[1] = (__bf16)a.y; x.h[2] = (__bf16)a.z; x.h[3] = (__bf16)a.w;
  x.h[4] = (__bf16)b.x; x.h[5] = (__bf16)b.y; x.h[6] = (__bf16)b.z; x.h[7] = (__bf16)b.w;
  *(uint4*)(out + i) = x.u;
}

// ---------------------------------------------------------------------------
// Projection GEMM (all-bf16 operands): C[m,n] = sum_k X[m,k]*W[n,k] + bias[n]
// M=4096, N=1024, K=1024. One wave per 16(M)x64(N) tile; A-frag reused 4x.
// MODE 0: out[b][h][s][d] (Q,K head-major)   MODE 1: out[b][h][d][s] (V^T)
// ---------------------------------------------------------------------------
template <int MODE>
__global__ __launch_bounds__(128)
void proj_kernel(const __bf16* __restrict__ X, const __bf16* __restrict__ W,
                 const float* __restrict__ bias, __bf16* __restrict__ out) {
  const int lane  = threadIdx.x & 31;
  const int wave  = threadIdx.x >> 5;
  const int tile  = blockIdx.x * 4 + wave;
  const int NT64  = HDIM / 64;
  const int m0    = (tile / NT64) * 16;
  const int n0    = (tile % NT64) * 64;
  const int half8 = (lane & 16) ? 8 : 0;

  const __bf16* arow = X + (size_t)(m0 + (lane & 15)) * HDIM;
  const __bf16* brow = W + (size_t)(n0 + (lane & 15)) * HDIM;
  v8f acc[4] = {};

  for (int k0 = 0; k0 < HDIM; k0 += 32) {
    const v16bf a = afrag_bf16(arow + k0);
#pragma unroll
    for (int t = 0; t < 4; ++t)
      acc[t] = wmma_bf16(a, bfrag_bf16(brow + (size_t)t * 16 * HDIM + k0), acc[t]);
  }

#pragma unroll
  for (int t = 0; t < 4; ++t) {
    const int n  = n0 + t * 16 + (lane & 15);
    const float bv = bias[n];
    const int h = n / DK, d = n % DK;
#pragma unroll
    for (int r = 0; r < 8; ++r) {
      const int m = m0 + r + half8;
      const int b = m / S_, s = m % S_;
      const float val = acc[t][r] + bv;
      size_t idx;
      if (MODE == 0) idx = (((size_t)(b * NH + h) * S_) + s) * DK + d;   // [b][h][s][d]
      else           idx = (((size_t)(b * NH + h) * DK) + d) * S_ + s;   // [b][h][d][s]
      out[idx] = (__bf16)val;
    }
  }
}

// ---------------------------------------------------------------------------
// Flash attention: one wave per 16-row q-tile of one (b,h), 64 keys per step.
//  - scores: 8 WMMAs (4 tiles x K=64)
//  - row max: 3 local v_max + 4-level shfl_xor per row (only cross-lane work)
//  - row sum: P x ones via 2 extra WMMAs into osum (rescaled like o tiles)
//  - P staged as bf16 in per-wave LDS tile (C-layout -> A-layout transpose)
//  - PV: 8 WMMAs into the 16x64 fp32 context accumulator
// ---------------------------------------------------------------------------
__global__ __launch_bounds__(128)
void attn_kernel(const __bf16* __restrict__ Qh, const __bf16* __restrict__ Kh,
                 const __bf16* __restrict__ Vt, const int* __restrict__ mask,
                 __bf16* __restrict__ CTX) {
  __shared__ __attribute__((aligned(16))) __bf16 pbuf[4][16 * PSTR];

  const int lane  = threadIdx.x & 31;
  const int wave  = threadIdx.x >> 5;
  const int half8 = (lane & 16) ? 8 : 0;
  const int bh    = blockIdx.y;
  const int b     = bh >> 4;
  const int h     = bh & 15;
  const int q0    = (blockIdx.x * 4 + wave) * 16;

  const __bf16* Qb = Qh + (size_t)bh * S_ * DK;
  const __bf16* Kb = Kh + (size_t)bh * S_ * DK;
  const __bf16* Vb = Vt + (size_t)bh * DK * S_;
  const int*    mb = mask + b * S_;
  __bf16*       pb = &pbuf[wave][0];

  const __bf16* qrow = Qb + (size_t)(q0 + (lane & 15)) * DK;
  const v16bf aq0 = afrag_bf16(qrow);
  const v16bf aq1 = afrag_bf16(qrow + 32);

  v16bf bones;                       // all-ones B fragment for row sums
#pragma unroll
  for (int i = 0; i < 16; ++i) bones[i] = (__bf16)1.0f;

  float mrow[8];
  v8f o[4] = {};
  v8f osum = {};
#pragma unroll
  for (int r = 0; r < 8; ++r) mrow[r] = -1e30f;

  for (int k0 = 0; k0 < S_; k0 += 64) {
    __builtin_prefetch(Kb + (size_t)(k0 + 64 + (lane & 15)) * DK, 0, 1);

    // ---- scores: four 16x16 tiles covering keys [k0, k0+64) ----------------
    v8f sc[4];
#pragma unroll
    for (int j = 0; j < 4; ++j) {
      const int kn = k0 + j * 16 + (lane & 15);
      const __bf16* krow = Kb + (size_t)kn * DK;
      v8f s = {};
      s = wmma_bf16(aq0, bfrag_bf16(krow), s);
      s = wmma_bf16(aq1, bfrag_bf16(krow + 32), s);
      const float madd = (mb[kn] == 0) ? -1e9f : 0.0f;  // mask folds into fma
#pragma unroll
      for (int r = 0; r < 8; ++r) s[r] = fmaf(s[r], 0.125f, madd);
      sc[j] = s;
    }

    // ---- online softmax: row max + exp; sums go through WMMA ---------------
#pragma unroll
    for (int r = 0; r < 8; ++r) {
      float t = fmaxf(fmaxf(sc[0][r], sc[1][r]), fmaxf(sc[2][r], sc[3][r]));
      t = fmaxf(t, __shfl_xor(t, 1));
      t = fmaxf(t, __shfl_xor(t, 2));
      t = fmaxf(t, __shfl_xor(t, 4));
      t = fmaxf(t, __shfl_xor(t, 8));
      const float mnew = fmaxf(mrow[r], t);
      const float corr = __expf(mrow[r] - mnew);
      mrow[r] = mnew;
      osum[r] *= corr;
#pragma unroll
      for (int t4 = 0; t4 < 4; ++t4) o[t4][r] *= corr;
      const int pr = (r + half8) * PSTR + (lane & 15);
#pragma unroll
      for (int j = 0; j < 4; ++j)
        pb[pr + j * 16] = (__bf16)__expf(sc[j][r] - mnew);
    }

    // per-wave LDS RAW: drain ds stores before the fragment reloads
    asm volatile("s_wait_dscnt 0" ::: "memory");

    // ---- P(16x64) x V(64x64) + rowsum(P) -----------------------------------
    const __bf16* prow = pb + (lane & 15) * PSTR;
    const v16bf ap0 = afrag_bf16(prow);        // keys k0   .. k0+31
    const v16bf ap1 = afrag_bf16(prow + 32);   // keys k0+32 .. k0+63
    osum = wmma_bf16(ap0, bones, osum);
    osum = wmma_bf16(ap1, bones, osum);
#pragma unroll
    for (int t4 = 0; t4 < 4; ++t4) {
      const __bf16* vrow = Vb + (size_t)(t4 * 16 + (lane & 15)) * S_ + k0;
      o[t4] = wmma_bf16(ap0, bfrag_bf16(vrow), o[t4]);
      o[t4] = wmma_bf16(ap1, bfrag_bf16(vrow + 32), o[t4]);
    }
  }

  // ---- epilogue: normalize by osum, scatter ctx to [b][s][h*64+d] ----------
#pragma unroll
  for (int r = 0; r < 8; ++r) {
    const float inv = 1.0f / osum[r];
    const int s = q0 + r + half8;
#pragma unroll
    for (int t4 = 0; t4 < 4; ++t4) {
      const int col = h * DK + t4 * 16 + (lane & 15);
      CTX[((size_t)b * S_ + s) * HDIM + col] = (__bf16)(o[t4][r] * inv);
    }
  }
}

// ---------------------------------------------------------------------------
// Output projection: out[m,n] = sum_k ctx[m,k]*Wo[n,k] + bo[n] (fp32 out)
// ---------------------------------------------------------------------------
__global__ __launch_bounds__(128)
void out_proj_kernel(const __bf16* __restrict__ CTX, const __bf16* __restrict__ Wo,
                     const float* __restrict__ bo, float* __restrict__ out) {
  const int lane  = threadIdx.x & 31;
  const int wave  = threadIdx.x >> 5;
  const int tile  = blockIdx.x * 4 + wave;
  const int NT64  = HDIM / 64;
  const int m0    = (tile / NT64) * 16;
  const int n0    = (tile % NT64) * 64;
  const int half8 = (lane & 16) ? 8 : 0;

  const __bf16* arow = CTX + (size_t)(m0 + (lane & 15)) * HDIM;
  const __bf16* brow = Wo + (size_t)(n0 + (lane & 15)) * HDIM;
  v8f acc[4] = {};

  for (int k0 = 0; k0 < HDIM; k0 += 32) {
    const v16bf a = afrag_bf16(arow + k0);
#pragma unroll
    for (int t = 0; t < 4; ++t)
      acc[t] = wmma_bf16(a, bfrag_bf16(brow + (size_t)t * 16 * HDIM + k0), acc[t]);
  }

#pragma unroll
  for (int t = 0; t < 4; ++t) {
    const int n = n0 + t * 16 + (lane & 15);
    const float bv = bo[n];
#pragma unroll
    for (int r = 0; r < 8; ++r) {
      const int m = m0 + r + half8;
      out[(size_t)m * HDIM + n] = acc[t][r] + bv;
    }
  }
}

// ---------------------------------------------------------------------------
extern "C" void kernel_launch(void* const* d_in, const int* in_sizes, int n_in,
                              void* d_out, int out_size, void* d_ws, size_t ws_size,
                              hipStream_t stream) {
  (void)in_sizes; (void)n_in; (void)out_size; (void)ws_size;
  const float* query = (const float*)d_in[0];
  const float* key   = (const float*)d_in[1];
  const float* value = (const float*)d_in[2];
  const int*   mask  = (const int*)d_in[3];
  const float* Wq = (const float*)d_in[4];  const float* bq = (const float*)d_in[5];
  const float* Wk = (const float*)d_in[6];  const float* bk = (const float*)d_in[7];
  const float* Wv = (const float*)d_in[8];  const float* bv = (const float*)d_in[9];
  const float* Wo = (const float*)d_in[10]; const float* bo = (const float*)d_in[11];
  float* out = (float*)d_out;

  const size_t MB = (size_t)1 << 20;
  char* ws = (char*)d_ws;
  __bf16* Qh  = (__bf16*)(ws + 0 * MB);    // [b][h][s][d]   8MB
  __bf16* Kh  = (__bf16*)(ws + 8 * MB);    // [b][h][s][d]   8MB
  __bf16* Vt  = (__bf16*)(ws + 16 * MB);   // [b][h][d][s]   8MB
  __bf16* Xq  = (__bf16*)(ws + 24 * MB);   // bf16 query     8MB (reused as Ctx)
  __bf16* Xk  = (__bf16*)(ws + 32 * MB);   // bf16 key       8MB
  __bf16* Xv  = (__bf16*)(ws + 40 * MB);   // bf16 value     8MB
  __bf16* Wqb = (__bf16*)(ws + 48 * MB);   // bf16 weights   2MB each
  __bf16* Wkb = (__bf16*)(ws + 50 * MB);
  __bf16* Wvb = (__bf16*)(ws + 52 * MB);
  __bf16* Wob = (__bf16*)(ws + 54 * MB);
  __bf16* Ctx = Xq;                        // query staging is dead by attn time

  const int n_x = B_ * S_ * HDIM;          // 4 Mi elements
  const int n_w = HDIM * HDIM;             // 1 Mi elements
  const dim3 cb(256);
  cvt_bf16_kernel<<<dim3(n_x / 8 / 256), cb, 0, stream>>>(query, Xq, n_x);
  cvt_bf16_kernel<<<dim3(n_x / 8 / 256), cb, 0, stream>>>(key,   Xk, n_x);
  cvt_bf16_kernel<<<dim3(n_x / 8 / 256), cb, 0, stream>>>(value, Xv, n_x);
  cvt_bf16_kernel<<<dim3(n_w / 8 / 256), cb, 0, stream>>>(Wq, Wqb, n_w);
  cvt_bf16_kernel<<<dim3(n_w / 8 / 256), cb, 0, stream>>>(Wk, Wkb, n_w);
  cvt_bf16_kernel<<<dim3(n_w / 8 / 256), cb, 0, stream>>>(Wv, Wvb, n_w);
  cvt_bf16_kernel<<<dim3(n_w / 8 / 256), cb, 0, stream>>>(Wo, Wob, n_w);

  const dim3 blk(128);                                   // 4 waves / block
  const dim3 gproj((B_ * S_ / 16) * (HDIM / 64) / 4);    // 1024 blocks
  proj_kernel<0><<<gproj, blk, 0, stream>>>(Xq, Wqb, bq, Qh);
  proj_kernel<0><<<gproj, blk, 0, stream>>>(Xk, Wkb, bk, Kh);
  proj_kernel<1><<<gproj, blk, 0, stream>>>(Xv, Wvb, bv, Vt);

  const dim3 gattn(S_ / 16 / 4, B_ * NH);                // (32, 32)
  attn_kernel<<<gattn, blk, 0, stream>>>(Qh, Kh, Vt, mask, Ctx);

  out_proj_kernel<<<gproj, blk, 0, stream>>>(Ctx, Wob, bo, out);
}
